// CA_SDGNN_87393994539671
// MI455X (gfx1250) — compile-verified, matrix operand order checked by reading, not run on hardware
//
#include <hip/hip_runtime.h>

typedef __attribute__((ext_vector_type(16))) _Float16 v16h;
typedef __attribute__((ext_vector_type(8)))  float    v8f;

#define Dd 64
#define Hh 4
#define Rr 4
#define Ll 2
#define HD 256    // H*D
#define QKVW 768  // fused Q|K|V row width

// ---------------- fills / copies / converts ----------------
__global__ void fill_i32(int* p, int v, int n) {
    int i = blockIdx.x * blockDim.x + threadIdx.x;
    if (i < n) p[i] = v;
}
__global__ void cvt_f32_f16(const float* __restrict__ in, _Float16* __restrict__ out, int n) {
    int i = blockIdx.x * blockDim.x + threadIdx.x;
    if (i < n) out[i] = (_Float16)in[i];
}
// W[K x Ncols] f32 -> Wt[Ncols x K] f16 (transposed for contiguous B-fragment loads)
__global__ void convT(const float* __restrict__ W, _Float16* __restrict__ Wt,
                      int K, int Ncols) {
    int i = blockIdx.x * blockDim.x + threadIdx.x;
    if (i >= K * Ncols) return;
    int k = i / Ncols, c = i - k * Ncols;
    Wt[(size_t)c * K + k] = (_Float16)W[i];
}
__global__ void concat_bias(const float* __restrict__ bq, const float* __restrict__ bk,
                            const float* __restrict__ bv, float* __restrict__ out) {
    int i = blockIdx.x * blockDim.x + threadIdx.x;
    if (i >= QKVW) return;
    out[i] = (i < HD) ? bq[i] : (i < 2 * HD) ? bk[i - HD] : bv[i - 2 * HD];
}

// ---------------- CSR build (once per relation; src is launch-invariant) -------------
__global__ void csr_count(const int* __restrict__ src, int* __restrict__ deg, int E) {
    int e = blockIdx.x * blockDim.x + threadIdx.x;
    if (e < E) atomicAdd(&deg[src[e]], 1);
}
// single-block exclusive scan over n (+ total at out[n])
__global__ void excl_scan_block(const int* __restrict__ in, int* __restrict__ out, int n) {
    __shared__ int sh[1024];
    __shared__ int carrySh;
    int tid = threadIdx.x;
    if (tid == 0) carrySh = 0;
    __syncthreads();
    for (int base = 0; base < n; base += 1024) {
        int v = (base + tid < n) ? in[base + tid] : 0;
        sh[tid] = v;
        __syncthreads();
        for (int off = 1; off < 1024; off <<= 1) {
            int t = (tid >= off) ? sh[tid - off] : 0;
            __syncthreads();
            sh[tid] += t;
            __syncthreads();
        }
        int carry = carrySh;
        if (base + tid < n) out[base + tid] = carry + sh[tid] - v;
        __syncthreads();
        if (tid == 0) carrySh = carry + sh[1023];
        __syncthreads();
    }
    if (tid == 0) out[n] = carrySh;
}
__global__ void csr_fill(const int* __restrict__ src, int* __restrict__ cursor,
                         int* __restrict__ perm, int E) {
    int e = blockIdx.x * blockDim.x + threadIdx.x;
    if (e < E) perm[atomicAdd(&cursor[src[e]], 1)] = e;
}
// canonicalize per-node edge order -> deterministic fp accumulation
__global__ void csr_sort(const int* __restrict__ offs, int* __restrict__ perm, int n) {
    int s = blockIdx.x * blockDim.x + threadIdx.x;
    if (s >= n) return;
    int beg = offs[s], end = offs[s + 1];
    for (int i = beg + 1; i < end; ++i) {
        int v = perm[i], j = i - 1;
        while (j >= beg && perm[j] > v) { perm[j + 1] = perm[j]; --j; }
        perm[j + 1] = v;
    }
}

// ---------------- generic WMMA GEMM: D = A @ Bt^T(f16) + bias + Cin ------------------
// A: [M x K] (f32 or f16, row-major). Bt: f16, column c's K-slice at Bt + c*ldb + bkoff.
// One wave computes a 16x16 tile; 8 waves per block.
template <typename AT>
__global__ void gemm_wmma(const AT* __restrict__ A, int lda,
                          const _Float16* __restrict__ Bt, int ldb, int bkoff,
                          const float* __restrict__ bias,
                          const float* __restrict__ Cin,
                          float* __restrict__ Dout, int ldd,
                          int M, int K) {
    int wave = threadIdx.x >> 5;
    int lane = threadIdx.x & 31;
    int row0 = (blockIdx.y * (blockDim.x >> 5) + wave) * 16;
    int col0 = blockIdx.x * 16;
    if (row0 >= M) return;

    int hi = lane >> 4;
    int lr = lane & 15;
    int kbase = hi * 8;    // lanes 0-15: K 0..7 & 16..23; lanes 16-31: K 8..15 & 24..31

    v8f acc = {};
    for (int k0 = 0; k0 < K; k0 += 32) {
        v16h a, b;
        const AT* Ap = A + (size_t)(row0 + lr) * lda + k0 + kbase;
        const _Float16* Bp = Bt + (size_t)(col0 + lr) * ldb + bkoff + k0 + kbase;
#pragma unroll
        for (int j = 0; j < 8; ++j) {
            a[j]     = (_Float16)Ap[j];
            a[j + 8] = (_Float16)Ap[j + 16];
            b[j]     = Bp[j];
            b[j + 8] = Bp[j + 16];
        }
        acc = __builtin_amdgcn_wmma_f32_16x16x32_f16(
            false, a, false, b, (short)0, acc, false, false);
    }

    int outCol = col0 + lr;
    float badd = bias ? bias[outCol] : 0.0f;
#pragma unroll
    for (int i = 0; i < 8; ++i) {
        int row = row0 + i + hi * 8;
        float v = acc[i] + badd;
        if (Cin) v += Cin[(size_t)row * ldd + outCol];
        Dout[(size_t)row * ldd + outCol] = v;
    }
}

// ---------------- centrality add ----------------
__global__ void cen_add(float* __restrict__ x, const int* __restrict__ nodes,
                        const float* __restrict__ betw, const float* __restrict__ clos,
                        const float* __restrict__ cenW, const float* __restrict__ cenb,
                        int n) {
    int i = blockIdx.x * blockDim.x + threadIdx.x;
    if (i >= n * Dd) return;
    int row = i >> 6, d = i & 63;
    int nd = nodes[row];
    x[i] += betw[nd] * cenW[d] + clos[nd] * cenW[Dd + d] + cenb[d];
}

// ---------------- edge scores (no atomics): scores[e,h] from fused QKV ----------------
__global__ void edge_scores(const float* __restrict__ qkv,
                            const int* __restrict__ src, const int* __restrict__ tgt,
                            const float* __restrict__ signInf,
                            const float* __restrict__ sw,   // sign_weight[l,:,r], stride R
                            float* __restrict__ scores, int Ecnt) {
    int i = blockIdx.x * blockDim.x + threadIdx.x;
    if (i >= Ecnt * Hh) return;
    int e = i >> 2, h = i & 3;
    int s = src[e], t = tgt[e];
    const float4* q = (const float4*)(qkv + (size_t)s * QKVW + h * Dd);
    const float4* k = (const float4*)(qkv + (size_t)t * QKVW + HD + h * Dd);
    float acc = 0.0f;
#pragma unroll
    for (int j = 0; j < 16; ++j) {
        float4 a = q[j], b = k[j];
        acc += a.x * b.x + a.y * b.y + a.z * b.z + a.w * b.w;
    }
    scores[(size_t)e * Hh + h] = acc * 0.125f * signInf[s] * sw[h * Rr];
}

// ---------------- per-node softmax over CSR edge list (wave per node) -----------------
__global__ void node_softmax(const int* __restrict__ offs, const int* __restrict__ perm,
                             float* __restrict__ scores, int n) {
    int lane = threadIdx.x & 31;
    int s = (blockIdx.x * blockDim.x + threadIdx.x) >> 5;
    if (s >= n) return;
    int beg = offs[s], end = offs[s + 1];
    float m0 = -3.4e38f, m1 = -3.4e38f, m2 = -3.4e38f, m3 = -3.4e38f;
    for (int i = beg + lane; i < end; i += 32) {
        float4 sc = *(const float4*)(scores + (size_t)perm[i] * Hh);
        m0 = fmaxf(m0, sc.x); m1 = fmaxf(m1, sc.y);
        m2 = fmaxf(m2, sc.z); m3 = fmaxf(m3, sc.w);
    }
#pragma unroll
    for (int o = 16; o > 0; o >>= 1) {
        m0 = fmaxf(m0, __shfl_xor(m0, o, 32)); m1 = fmaxf(m1, __shfl_xor(m1, o, 32));
        m2 = fmaxf(m2, __shfl_xor(m2, o, 32)); m3 = fmaxf(m3, __shfl_xor(m3, o, 32));
    }
    float z0 = 0, z1 = 0, z2 = 0, z3 = 0;
    for (int i = beg + lane; i < end; i += 32) {
        float4 sc = *(const float4*)(scores + (size_t)perm[i] * Hh);
        z0 += __expf(sc.x - m0); z1 += __expf(sc.y - m1);
        z2 += __expf(sc.z - m2); z3 += __expf(sc.w - m3);
    }
#pragma unroll
    for (int o = 16; o > 0; o >>= 1) {
        z0 += __shfl_xor(z0, o, 32); z1 += __shfl_xor(z1, o, 32);
        z2 += __shfl_xor(z2, o, 32); z3 += __shfl_xor(z3, o, 32);
    }
    float i0 = 1.0f / (z0 + 1e-10f), i1 = 1.0f / (z1 + 1e-10f);
    float i2 = 1.0f / (z2 + 1e-10f), i3 = 1.0f / (z3 + 1e-10f);
    for (int i = beg + lane; i < end; i += 32) {
        float4* p = (float4*)(scores + (size_t)perm[i] * Hh);
        float4 sc = *p;
        sc.x = __expf(sc.x - m0) * i0; sc.y = __expf(sc.y - m1) * i1;
        sc.z = __expf(sc.z - m2) * i2; sc.w = __expf(sc.w - m3) * i3;
        *p = sc;   // buffer now holds attention weights
    }
}

// ---------------- per-node aggregation (wave per node, no atomics) --------------------
// lane owns 8 contiguous outputs [lane*8 .. lane*8+8): head = lane>>3, b128 V loads.
__global__ void node_aggregate(const int* __restrict__ offs, const int* __restrict__ perm,
                               const int* __restrict__ tgt,
                               const float* __restrict__ attn,   // [E,4]
                               const float* __restrict__ qkv,
                               float* __restrict__ outb, int n) {
    int lane = threadIdx.x & 31;
    int s = (blockIdx.x * blockDim.x + threadIdx.x) >> 5;
    if (s >= n) return;
    int beg = offs[s], end = offs[s + 1];
    int h = lane >> 3;
    float acc[8] = {0, 0, 0, 0, 0, 0, 0, 0};
    for (int i = beg; i < end; ++i) {
        int e = perm[i];
        int t = tgt[e];
        float a = attn[(size_t)e * Hh + h];          // lane-group broadcast
        const float4* v = (const float4*)(qkv + (size_t)t * QKVW + 2 * HD + lane * 8);
        float4 v0 = v[0], v1 = v[1];
        acc[0] += v0.x * a; acc[1] += v0.y * a; acc[2] += v0.z * a; acc[3] += v0.w * a;
        acc[4] += v1.x * a; acc[5] += v1.y * a; acc[6] += v1.z * a; acc[7] += v1.w * a;
    }
    float4* o = (float4*)(outb + (size_t)s * HD + lane * 8);
    o[0] = make_float4(acc[0], acc[1], acc[2], acc[3]);
    o[1] = make_float4(acc[4], acc[5], acc[6], acc[7]);
}

// ---------------- residual + LayerNorm (wave per row of 64) ----------------
__global__ void ln_kernel(const float* __restrict__ x, const float* __restrict__ acc,
                          const float* __restrict__ gamma, const float* __restrict__ beta,
                          float* __restrict__ xout, int n) {
    int lane = threadIdx.x & 31;
    int row = (blockIdx.x * blockDim.x + threadIdx.x) >> 5;
    if (row >= n) return;
    size_t base = (size_t)row * Dd;
    float y0 = x[base + lane]      + acc[base + lane];
    float y1 = x[base + 32 + lane] + acc[base + 32 + lane];
    float ssum = y0 + y1;
#pragma unroll
    for (int o = 16; o > 0; o >>= 1) ssum += __shfl_xor(ssum, o, 32);
    float mean = ssum * (1.0f / 64.0f);
    float d0 = y0 - mean, d1 = y1 - mean;
    float vs = d0 * d0 + d1 * d1;
#pragma unroll
    for (int o = 16; o > 0; o >>= 1) vs += __shfl_xor(vs, o, 32);
    float inv = __frsqrt_rn(vs * (1.0f / 64.0f) + 1e-5f);
    xout[base + lane]      = d0 * inv * gamma[lane]      + beta[lane];
    xout[base + 32 + lane] = d1 * inv * gamma[32 + lane] + beta[32 + lane];
}

// ---------------- host side ----------------
static inline void* wsTake(char*& p, size_t bytes) {
    void* r = (void*)p;
    p += (bytes + 255) & ~(size_t)255;
    return r;
}

extern "C" void kernel_launch(void* const* d_in, const int* in_sizes, int n_in,
                              void* d_out, int out_size, void* d_ws, size_t ws_size,
                              hipStream_t stream) {
    const int*   nodes   = (const int*)d_in[0];
    const float* nodeF   = (const float*)d_in[1];
    const float* betw    = (const float*)d_in[2];
    const float* clos    = (const float*)d_in[3];
    const float* signInf = (const float*)d_in[4];
    const int*   src     = (const int*)d_in[5];
    const int*   tgt     = (const int*)d_in[6];
    const float* enc_W   = (const float*)d_in[7];
    const float* enc_b   = (const float*)d_in[8];
    const float* cen_W   = (const float*)d_in[9];
    const float* cen_b   = (const float*)d_in[10];
    const float* Wq      = (const float*)d_in[11];
    const float* bq      = (const float*)d_in[12];
    const float* Wk      = (const float*)d_in[13];
    const float* bk      = (const float*)d_in[14];
    const float* Wv      = (const float*)d_in[15];
    const float* bv      = (const float*)d_in[16];
    const float* signW   = (const float*)d_in[17];
    const float* projW   = (const float*)d_in[18];
    const float* projB   = (const float*)d_in[19];
    const float* lnG     = (const float*)d_in[20];
    const float* lnB     = (const float*)d_in[21];

    const int N = in_sizes[0];
    const int E = in_sizes[5] / Rr;

    // workspace carve-out (~107 MB)
    char* p = (char*)d_ws;
    float*    x       = (float*)wsTake(p, (size_t)N * Dd * 4);
    _Float16* xh      = (_Float16*)wsTake(p, (size_t)N * Dd * 2);
    float*    accP    = (float*)wsTake(p, (size_t)N * Dd * 4);
    float*    qkv     = (float*)wsTake(p, (size_t)N * QKVW * 4);
    float*    outb    = (float*)wsTake(p, (size_t)N * HD * 4);
    float*    scores  = (float*)wsTake(p, (size_t)E * Hh * 4);
    int*      perm    = (int*)wsTake(p, (size_t)Rr * E * 4);
    int*      offs    = (int*)wsTake(p, (size_t)Rr * (N + 1) * 4);
    int*      deg     = (int*)wsTake(p, (size_t)N * 4);
    int*      cursor  = (int*)wsTake(p, (size_t)N * 4);
    _Float16* encWt   = (_Float16*)wsTake(p, (size_t)Dd * Dd * 2);
    _Float16* qkvWt   = (_Float16*)wsTake(p, (size_t)3 * Ll * Rr * Dd * HD * 2);
    _Float16* projWt  = (_Float16*)wsTake(p, (size_t)Ll * HD * Rr * Dd * 2);
    float*    qkvBias = (float*)wsTake(p, (size_t)Ll * Rr * QKVW * 4);
    (void)ws_size; (void)n_in; (void)out_size;

    const int TB = 256;
    auto blocks = [](long long n, int tb) { return (unsigned)((n + tb - 1) / tb); };

    // ---- weights -> transposed f16; fused QKV biases ----
    convT<<<blocks((long long)Dd * Dd, TB), TB, 0, stream>>>(enc_W, encWt, Dd, Dd);
    for (int l = 0; l < Ll; ++l)
        for (int r = 0; r < Rr; ++r) {
            size_t wo = (size_t)(l * Rr + r) * Dd * HD;
            size_t to = (size_t)((l * Rr + r) * 3) * Dd * HD;
            convT<<<blocks((long long)Dd * HD, TB), TB, 0, stream>>>(Wq + wo, qkvWt + to,                   Dd, HD);
            convT<<<blocks((long long)Dd * HD, TB), TB, 0, stream>>>(Wk + wo, qkvWt + to + (size_t)Dd * HD,  Dd, HD);
            convT<<<blocks((long long)Dd * HD, TB), TB, 0, stream>>>(Wv + wo, qkvWt + to + (size_t)2 * Dd * HD, Dd, HD);
            size_t bo = (size_t)(l * Rr + r) * HD;
            concat_bias<<<blocks(QKVW, TB), TB, 0, stream>>>(bq + bo, bk + bo, bv + bo,
                                                             qkvBias + (size_t)(l * Rr + r) * QKVW);
        }
    for (int l = 0; l < Ll; ++l)
        convT<<<blocks((long long)HD * Rr * Dd, TB), TB, 0, stream>>>(
            projW + (size_t)l * HD * Rr * Dd, projWt + (size_t)l * HD * Rr * Dd, HD * Rr, Dd);

    // ---- CSR per relation (deterministic: per-node sorted edge ids) ----
    for (int r = 0; r < Rr; ++r) {
        const int* sr = src + (size_t)r * E;
        int* offr = offs + (size_t)r * (N + 1);
        int* permr = perm + (size_t)r * E;
        fill_i32<<<blocks(N, TB), TB, 0, stream>>>(deg, 0, N);
        csr_count<<<blocks(E, TB), TB, 0, stream>>>(sr, deg, E);
        excl_scan_block<<<1, 1024, 0, stream>>>(deg, offr, N);
        hipMemcpyAsync(cursor, offr, (size_t)N * 4, hipMemcpyDeviceToDevice, stream);
        csr_fill<<<blocks(E, TB), TB, 0, stream>>>(sr, cursor, permr, E);
        csr_sort<<<blocks(N, TB), TB, 0, stream>>>(offr, permr, N);
    }

    const int mtiles = (N + 15) / 16;
    const dim3 gGemm64((unsigned)(Dd / 16),   blocks(mtiles, 8));
    const dim3 gGemm768((unsigned)(QKVW / 16), blocks(mtiles, 8));

    // ---- encoder ----
    gemm_wmma<float><<<gGemm64, TB, 0, stream>>>(nodeF, Dd, encWt, Dd, 0,
                                                 enc_b, nullptr, x, Dd, N, Dd);
    cen_add<<<blocks((long long)N * Dd, TB), TB, 0, stream>>>(x, nodes, betw, clos,
                                                              cen_W, cen_b, N);
    cvt_f32_f16<<<blocks((long long)N * Dd, TB), TB, 0, stream>>>(x, xh, N * Dd);

    // ---- layers ----
    for (int l = 0; l < Ll; ++l) {
        for (int r = 0; r < Rr; ++r) {
            size_t to = (size_t)((l * Rr + r) * 3) * Dd * HD;
            const int* sr = src + (size_t)r * E;
            const int* tr = tgt + (size_t)r * E;
            const int* offr = offs + (size_t)r * (N + 1);
            const int* permr = perm + (size_t)r * E;
            // fused QKV GEMM: [N x 64](f16) @ [64 x 768] -> qkv[N x 768]
            gemm_wmma<_Float16><<<gGemm768, TB, 0, stream>>>(
                xh, Dd, qkvWt + to, Dd, 0,
                qkvBias + (size_t)(l * Rr + r) * QKVW, nullptr, qkv, QKVW, N, Dd);
            // edge scores, per-node softmax, per-node aggregation (all atomic-free)
            edge_scores<<<blocks((long long)E * Hh, TB), TB, 0, stream>>>(
                qkv, sr, tr, signInf, signW + l * Hh * Rr + r, scores, E);
            node_softmax<<<blocks((long long)N * 32, TB), TB, 0, stream>>>(
                offr, permr, scores, N);
            node_aggregate<<<blocks((long long)N * 32, TB), TB, 0, stream>>>(
                offr, permr, tr, scores, qkv, outb, N);
            // projection slice accumulate: accP (+)= outb @ proj_W[l, r*256:(r+1)*256, :]
            gemm_wmma<float><<<gGemm64, TB, 0, stream>>>(
                outb, HD, projWt + (size_t)l * HD * Rr * Dd, HD * Rr, r * HD,
                (r == 0) ? projB + l * Dd : nullptr,
                (r == 0) ? nullptr : accP,
                accP, Dd, N, HD);
        }
        // residual + LayerNorm (final layer writes straight to d_out)
        float* xout = (l == Ll - 1) ? (float*)d_out : x;
        ln_kernel<<<blocks((long long)N * 32, TB), TB, 0, stream>>>(
            x, accP, lnG + l * Dd, lnB + l * Dd, xout, N);
        if (l != Ll - 1)
            cvt_f32_f16<<<blocks((long long)N * Dd, TB), TB, 0, stream>>>(x, xh, N * Dd);
    }
}
